// KAFPGNN_76871324663925
// MI455X (gfx1250) — matrix operand order repaired and checked by reading, not compiled, not run on hardware
//
#include <hip/hip_runtime.h>
#include <math.h>

// ---------------------------------------------------------------------------
// Types for CDNA5 WMMA (wave32): v_wmma_f32_16x16x32_bf16
// ---------------------------------------------------------------------------
typedef __attribute__((ext_vector_type(16))) __bf16 v16bf;
typedef __attribute__((ext_vector_type(8)))  __bf16 v8bf;
typedef __attribute__((ext_vector_type(8)))  float  v8f;

#define DD   256   // node feat dim
#define KP   20    // kron proj dim
#define KK   400   // KP*KP
#define KKP  416   // padded to 13 * 32
#define OUTD 256

static __device__ __forceinline__ v16bf mk16(v8bf lo, v8bf hi) {
  return __builtin_shufflevector(lo, hi, 0,1,2,3,4,5,6,7,8,9,10,11,12,13,14,15);
}
static __device__ __forceinline__ v8f wmma_bf16(v16bf a, v16bf b, v8f c) {
  return __builtin_amdgcn_wmma_f32_16x16x32_bf16(false, a, false, b, (short)0, c,
                                                 false, false);
}
static __device__ __forceinline__ float wred16(float v) {
  #pragma unroll
  for (int o = 1; o < 16; o <<= 1) v += __shfl_xor(v, o, 32);
  return v;
}
static __device__ __forceinline__ float wred32(float v) {
  #pragma unroll
  for (int o = 1; o < 32; o <<= 1) v += __shfl_xor(v, o, 32);
  return v;
}
static __device__ __forceinline__ float sigmoidf_(float x) {
  return 1.0f / (1.0f + expf(-x));
}

// ---------------------------------------------------------------------------
// Utility kernels
// ---------------------------------------------------------------------------
__global__ void k_zero_f32(float* p, long long n) {
  long long i = (long long)blockIdx.x * blockDim.x + threadIdx.x;
  if (i < n) p[i] = 0.0f;
}
__global__ void k_cvt_bf16(const float* __restrict__ x, __bf16* __restrict__ y,
                           long long n) {
  long long i = (long long)blockIdx.x * blockDim.x + threadIdx.x;
  if (i < n) y[i] = (__bf16)x[i];
}
__global__ void k_relu_cvt_bf16(const float* __restrict__ x,
                                __bf16* __restrict__ y, long long n) {
  long long i = (long long)blockIdx.x * blockDim.x + threadIdx.x;
  if (i < n) y[i] = (__bf16)fmaxf(x[i], 0.0f);
}

// Pack a [K,N] (or transposed [N,K]) fp32 weight into WMMA B-fragment order:
// out[((kstep*(N/16)+ntile)*32 + lane)*16 + el] = B[kstep*32 + (el>=8?16:0) + hi*8 + (el&7)][n]
// where hi = lane>>4, n = ntile*16 + (lane&15). Zero-pad K -> Kpad.
__global__ void k_pack_b(const float* __restrict__ W, __bf16* __restrict__ out,
                         int K, int Kpad, int N, int trans) {
  int ntiles = N >> 4;
  long long total = (long long)(Kpad >> 5) * ntiles * 512;
  long long i = (long long)blockIdx.x * blockDim.x + threadIdx.x;
  if (i >= total) return;
  int r = (int)(i & 511);
  long long tile = i >> 9;
  int ntile = (int)(tile % ntiles);
  int kstep = (int)(tile / ntiles);
  int lane = r >> 4, el = r & 15;
  int hi = lane >> 4, lr = lane & 15;
  int n = ntile * 16 + lr;
  int k = kstep * 32 + (el >= 8 ? 16 : 0) + hi * 8 + (el & 7);
  float v = 0.0f;
  if (k < K) v = trans ? W[(long long)n * K + k] : W[(long long)k * N + n];
  out[i] = (__bf16)v;
}

// ---------------------------------------------------------------------------
// npj = relu(LN(node_feats @ Wp + bp))  -> [V, 20] fp32
// block: 320 threads = 16 nodes x 20 columns
// ---------------------------------------------------------------------------
__global__ __launch_bounds__(320)
void k_node_proj(const float* __restrict__ nf, const float* __restrict__ Wp,
                 const float* __restrict__ bp, const float* __restrict__ gp,
                 const float* __restrict__ betap, float* __restrict__ npj,
                 int V) {
  __shared__ float s_x[16][KP];
  __shared__ float s_mu[16], s_rv[16];
  int tid = threadIdx.x;
  int m = tid / KP, a = tid % KP;
  int v = blockIdx.x * 16 + m;
  float acc = 0.0f;
  if (tid < 16 * KP && v < V) {
    const float* row = nf + (long long)v * DD;
    for (int d = 0; d < DD; ++d) acc += row[d] * Wp[d * KP + a];
    acc += bp[a];
    s_x[m][a] = acc;
  }
  __syncthreads();
  if (tid < 16 && blockIdx.x * 16 + tid < V) {
    float s = 0.f, sq = 0.f;
    for (int j = 0; j < KP; ++j) { float x = s_x[tid][j]; s += x; sq += x * x; }
    float mu = s / KP;
    float var = sq / KP - mu * mu;
    s_mu[tid] = mu;
    s_rv[tid] = rsqrtf(var + 1e-5f);
  }
  __syncthreads();
  if (tid < 16 * KP && v < V) {
    float y = (acc - s_mu[m]) * s_rv[m] * gp[a] + betap[a];
    npj[(long long)v * KP + a] = fmaxf(y, 0.0f);
  }
}

// ---------------------------------------------------------------------------
// Fused Kronecker edge GEMM + LayerNorm + ReLU + segment-sum scatter.
// block = 256 threads, 16 edges. A (16x416 bf16) built in A-fragment order in
// LDS from outer(npj[src], npj[dst]); 8 waves x 2 ntiles of WMMA; then LN
// across the 256 outputs per edge; atomic scatter to kron_feat[dst].
// ---------------------------------------------------------------------------
__global__ __launch_bounds__(256)
void k_edge_kron(const float* __restrict__ npj, const int* __restrict__ src,
                 const int* __restrict__ dst, const __bf16* __restrict__ WkB,
                 const float* __restrict__ bk, const float* __restrict__ gk,
                 const float* __restrict__ betak, float* __restrict__ kron_feat,
                 int E) {
  __shared__ float s_nps[16][KP];
  __shared__ float s_npd[16][KP];
  __shared__ __bf16 s_afrag[13 * 32 * 16];
  __shared__ int s_dst[16];
  __shared__ float s_sum[16], s_sq[16], s_mu[16], s_rv[16];

  const int tid = threadIdx.x;
  const int e0 = blockIdx.x * 16;
  if (tid < 16) {
    int e = min(e0 + tid, E - 1);
    s_dst[tid] = dst[e];
    s_sum[tid] = 0.0f;
    s_sq[tid]  = 0.0f;
  }
  for (int i = tid; i < 16 * KP; i += 256) {
    int m = i / KP, a = i % KP;
    int e = min(e0 + m, E - 1);
    s_nps[m][a] = npj[(long long)src[e] * KP + a];
    s_npd[m][a] = npj[(long long)dst[e] * KP + a];
  }
  __syncthreads();

  // Build A fragments: afrag[(ks*32 + lane)*16 + el]
  for (int i = tid; i < 13 * 512; i += 256) {
    int ks = i >> 9;
    int r = i & 511;
    int lane = r >> 4, el = r & 15;
    int m = lane & 15, hi = lane >> 4;
    int kk = (el >= 8 ? 16 : 0) + hi * 8 + (el & 7);
    int j = ks * 32 + kk;                       // column of kron row
    float v = 0.0f;
    if (j < KK) v = s_nps[m][j / KP] * s_npd[m][j % KP];
    s_afrag[i] = (__bf16)v;
  }
  __syncthreads();

  const int wid = tid >> 5, lane = tid & 31;
  const int hi = lane >> 4, lr = lane & 15;
  v8f acc0 = {0.f,0.f,0.f,0.f,0.f,0.f,0.f,0.f};
  v8f acc1 = acc0;
  const int nt0 = wid * 2, nt1 = wid * 2 + 1;
  #pragma unroll
  for (int ks = 0; ks < 13; ++ks) {
    const v8bf* ap = (const v8bf*)&s_afrag[(ks * 32 + lane) * 16];
    v16bf a = mk16(ap[0], ap[1]);
    const __bf16* b0p = WkB + ((long long)(ks * 16 + nt0) * 32 + lane) * 16;
    const __bf16* b1p = WkB + ((long long)(ks * 16 + nt1) * 32 + lane) * 16;
    v16bf b0 = mk16(*(const v8bf*)b0p, *(const v8bf*)(b0p + 8));
    v16bf b1 = mk16(*(const v8bf*)b1p, *(const v8bf*)(b1p + 8));
    acc0 = wmma_bf16(a, b0, acc0);
    acc1 = wmma_bf16(a, b1, acc1);
  }

  // bias + per-edge LayerNorm statistics over all 256 cols (across 8 waves)
  const int n0 = nt0 * 16 + lr;
  const int n1 = nt1 * 16 + lr;
  const float bk0 = bk[n0], bk1 = bk[n1];
  float x0[8], x1[8];
  #pragma unroll
  for (int r = 0; r < 8; ++r) { x0[r] = acc0[r] + bk0; x1[r] = acc1[r] + bk1; }
  #pragma unroll
  for (int r = 0; r < 8; ++r) {
    float sm = x0[r] + x1[r];
    float sq = x0[r] * x0[r] + x1[r] * x1[r];
    sm = wred16(sm);
    sq = wred16(sq);
    if (lr == 0) {
      int m = r + hi * 8;
      atomicAdd(&s_sum[m], sm);
      atomicAdd(&s_sq[m], sq);
    }
  }
  __syncthreads();
  if (tid < 16) {
    float mu = s_sum[tid] * (1.0f / 256.0f);
    float var = s_sq[tid] * (1.0f / 256.0f) - mu * mu;
    s_mu[tid] = mu;
    s_rv[tid] = rsqrtf(var + 1e-5f);
  }
  __syncthreads();

  const float g0 = gk[n0], g1 = gk[n1];
  const float bb0 = betak[n0], bb1 = betak[n1];
  #pragma unroll
  for (int r = 0; r < 8; ++r) {
    int m = r + hi * 8;
    if (e0 + m < E) {
      float mu = s_mu[m], rv = s_rv[m];
      long long base = (long long)s_dst[m] * OUTD;
      float y0 = fmaxf((x0[r] - mu) * rv * g0 + bb0, 0.0f);
      float y1 = fmaxf((x1[r] - mu) * rv * g1 + bb1, 0.0f);
      atomicAdd(&kron_feat[base + n0], y0);
      atomicAdd(&kron_feat[base + n1], y1);
    }
  }
}

// ---------------------------------------------------------------------------
// Generic bf16 WMMA GEMM: C[M,N] = A[M,K] @ Bfrag + bias (fp32 out).
// One block (256 thr, 8 waves) per 16-row tile; each wave owns NT n-tiles.
// Requires N == 8*NT*16, K % 32 == 0, M % 16 == 0.
// ---------------------------------------------------------------------------
template <int NT>
__global__ __launch_bounds__(256)
void k_gemm_bf16(const __bf16* __restrict__ A, int K,
                 const __bf16* __restrict__ Bfrag, int N,
                 const float* __restrict__ bias, float* __restrict__ C) {
  const int tid = threadIdx.x;
  const int wid = tid >> 5, lane = tid & 31;
  const int hi = lane >> 4, lr = lane & 15;
  const int ntiles = N >> 4;
  const int ksteps = K >> 5;

  v8f acc[NT];
  #pragma unroll
  for (int t = 0; t < NT; ++t) {
    v8f z = {0.f,0.f,0.f,0.f,0.f,0.f,0.f,0.f};
    acc[t] = z;
  }

  const __bf16* Arow = A + (long long)(blockIdx.x * 16 + lr) * K;
  for (int ks = 0; ks < ksteps; ++ks) {
    int k0 = ks * 32 + hi * 8;
    v16bf a = mk16(*(const v8bf*)(Arow + k0), *(const v8bf*)(Arow + k0 + 16));
    #pragma unroll
    for (int t = 0; t < NT; ++t) {
      int nt = wid * NT + t;
      const __bf16* bp = Bfrag + ((long long)(ks * ntiles + nt) * 32 + lane) * 16;
      v16bf b = mk16(*(const v8bf*)bp, *(const v8bf*)(bp + 8));
      acc[t] = wmma_bf16(a, b, acc[t]);
    }
  }

  #pragma unroll
  for (int t = 0; t < NT; ++t) {
    int n = (wid * NT + t) * 16 + lr;
    float bv = bias[n];
    #pragma unroll
    for (int r = 0; r < 8; ++r) {
      long long m = (long long)blockIdx.x * 16 + r + hi * 8;
      C[m * N + n] = acc[t][r] + bv;
    }
  }
}

// ---------------------------------------------------------------------------
// Attention path
// ---------------------------------------------------------------------------
// s1[v] = nf[v].We[0:256], s2[v] = nf[v].We[256:512]   (wave per node)
__global__ __launch_bounds__(256)
void k_node_s12(const float* __restrict__ nf, const float* __restrict__ We,
                float* __restrict__ s1, float* __restrict__ s2, int V) {
  int wgl = (blockIdx.x * blockDim.x + threadIdx.x) >> 5;
  int lane = threadIdx.x & 31;
  if (wgl >= V) return;
  const float* row = nf + (long long)wgl * DD;
  float a = 0.f, b = 0.f;
  #pragma unroll
  for (int c = 0; c < 8; ++c) {
    int d = c * 32 + lane;
    float x = row[d];
    a += x * We[d];
    b += x * We[DD + d];
  }
  a = wred32(a);
  b = wred32(b);
  if (lane == 0) { s1[wgl] = a; s2[wgl] = b; }
}

__global__ void k_edge_logit_max(const float* __restrict__ s1,
                                 const float* __restrict__ s2,
                                 const float* __restrict__ be,
                                 const int* __restrict__ src,
                                 const int* __restrict__ dst,
                                 float* __restrict__ logit,
                                 float* __restrict__ mmax, int E) {
  int e = blockIdx.x * blockDim.x + threadIdx.x;
  if (e >= E) return;
  float l = fmaxf(s1[dst[e]] + s2[src[e]] + be[0], 0.0f);
  logit[e] = l;
  atomicMax((int*)&mmax[dst[e]], __float_as_int(l));  // l >= 0, init 0
}

__global__ void k_edge_exp_sum(const float* __restrict__ logit,
                               const float* __restrict__ mmax,
                               const int* __restrict__ dst,
                               float* __restrict__ ex,
                               float* __restrict__ denom, int E) {
  int e = blockIdx.x * blockDim.x + threadIdx.x;
  if (e >= E) return;
  float v = expf(logit[e] - mmax[dst[e]]);
  ex[e] = v;
  atomicAdd(&denom[dst[e]], v);
}

__global__ void k_edge_att(const float* __restrict__ ex,
                           const float* __restrict__ denom,
                           const int* __restrict__ dst,
                           float* __restrict__ att, int E) {
  int e = blockIdx.x * blockDim.x + threadIdx.x;
  if (e >= E) return;
  att[e] = ex[e] / denom[dst[e]];
}

// context[dst] += att[e] * hv[src]  (64 threads/edge, float4 each)
__global__ void k_context_agg(const float* __restrict__ hv,
                              const float* __restrict__ att,
                              const int* __restrict__ src,
                              const int* __restrict__ dst,
                              float* __restrict__ context, long long E) {
  long long gid = (long long)blockIdx.x * blockDim.x + threadIdx.x;
  if (gid >= E * 64) return;
  long long e = gid >> 6;
  int c = (int)(gid & 63);
  int d = c * 4;
  float a = att[e];
  const float4 h = *(const float4*)(hv + (long long)src[e] * DD + d);
  float* out = context + (long long)dst[e] * DD + d;
  atomicAdd(out + 0, a * h.x);
  atomicAdd(out + 1, a * h.y);
  atomicAdd(out + 2, a * h.z);
  atomicAdd(out + 3, a * h.w);
}

// ---------------------------------------------------------------------------
// GRU gates + LN; writes cat_bf16[v] = [LN(relu(h)), kron_feat[v]] (bf16)
// wave per node; gi/gh are chunk-local buffers.
// ---------------------------------------------------------------------------
__global__ __launch_bounds__(256)
void k_gru_gates(const float* __restrict__ gi, const float* __restrict__ gh,
                 const float* __restrict__ nf, const float* __restrict__ kron_feat,
                 const float* __restrict__ g_ln, const float* __restrict__ b_ln,
                 __bf16* __restrict__ catb, int nodeBase, int numNodes) {
  int nloc = blockIdx.x * (blockDim.x >> 5) + (threadIdx.x >> 5);
  int lane = threadIdx.x & 31;
  if (nloc >= numNodes) return;
  long long v = (long long)nodeBase + nloc;
  const float* giR = gi + (long long)nloc * 768;
  const float* ghR = gh + (long long)nloc * 768;
  const float* nfR = nf + v * DD;

  float x[8];
  float s = 0.f, sq = 0.f;
  #pragma unroll
  for (int c = 0; c < 8; ++c) {
    int d = c * 32 + lane;
    float ir = giR[d],       hr = ghR[d];
    float iz = giR[256 + d], hz = ghR[256 + d];
    float in_ = giR[512 + d], hn = ghR[512 + d];
    float r = sigmoidf_(ir + hr);
    float z = sigmoidf_(iz + hz);
    float n = tanhf(in_ + r * hn);
    float h = (1.0f - z) * n + z * nfR[d];
    float xx = fmaxf(h, 0.0f);
    x[c] = xx;
    s += xx;
    sq += xx * xx;
  }
  s = wred32(s);
  sq = wred32(sq);
  float mu = s * (1.0f / 256.0f);
  float rv = rsqrtf(sq * (1.0f / 256.0f) - mu * mu + 1e-5f);
  __bf16* cR = catb + v * 512;
  const float* kfR = kron_feat + v * OUTD;
  #pragma unroll
  for (int c = 0; c < 8; ++c) {
    int d = c * 32 + lane;
    float y = (x[c] - mu) * rv * g_ln[d] + b_ln[d];
    cR[d] = (__bf16)y;
    cR[256 + d] = (__bf16)kfR[d];
  }
}

// final out = relu(LN(tmp)) with gc, betac  (wave per node)
__global__ __launch_bounds__(256)
void k_ln_relu_out(const float* __restrict__ x, const float* __restrict__ g,
                   const float* __restrict__ b, float* __restrict__ out, int V) {
  int v = blockIdx.x * (blockDim.x >> 5) + (threadIdx.x >> 5);
  int lane = threadIdx.x & 31;
  if (v >= V) return;
  const float* xR = x + (long long)v * DD;
  float xl[8];
  float s = 0.f, sq = 0.f;
  #pragma unroll
  for (int c = 0; c < 8; ++c) {
    int d = c * 32 + lane;
    float t = xR[d];
    xl[c] = t;
    s += t;
    sq += t * t;
  }
  s = wred32(s);
  sq = wred32(sq);
  float mu = s * (1.0f / 256.0f);
  float rv = rsqrtf(sq * (1.0f / 256.0f) - mu * mu + 1e-5f);
  float* oR = out + (long long)v * DD;
  #pragma unroll
  for (int c = 0; c < 8; ++c) {
    int d = c * 32 + lane;
    oR[d] = fmaxf((xl[c] - mu) * rv * g[d] + b[d], 0.0f);
  }
}

// ---------------------------------------------------------------------------
// Host launcher
// ---------------------------------------------------------------------------
extern "C" void kernel_launch(void* const* d_in, const int* in_sizes, int n_in,
                              void* d_out, int out_size, void* d_ws,
                              size_t ws_size, hipStream_t stream) {
  const float* nf    = (const float*)d_in[0];
  const int*   src   = (const int*)d_in[1];
  const int*   dst   = (const int*)d_in[2];
  const float* Wp    = (const float*)d_in[3];
  const float* bp    = (const float*)d_in[4];
  const float* gp    = (const float*)d_in[5];
  const float* betap = (const float*)d_in[6];
  const float* Wk    = (const float*)d_in[7];
  const float* bk    = (const float*)d_in[8];
  const float* gk    = (const float*)d_in[9];
  const float* betak = (const float*)d_in[10];
  const float* We    = (const float*)d_in[11];
  const float* be    = (const float*)d_in[12];
  const float* Wn    = (const float*)d_in[13];
  const float* bn    = (const float*)d_in[14];
  const float* W_ih  = (const float*)d_in[15];
  const float* W_hh  = (const float*)d_in[16];
  const float* b_ih  = (const float*)d_in[17];
  const float* b_hh  = (const float*)d_in[18];
  const float* g_ln  = (const float*)d_in[19];
  const float* b_ln  = (const float*)d_in[20];
  const float* Wc    = (const float*)d_in[21];
  const float* bc    = (const float*)d_in[22];
  const float* gc    = (const float*)d_in[23];
  const float* betac = (const float*)d_in[24];

  const int V = in_sizes[0] / DD;
  const int E = in_sizes[1];

  // ---- workspace sub-allocation (256B aligned) ----
  char* ws = (char*)d_ws;
  size_t off = 0;
  auto alloc = [&](size_t bytes) -> char* {
    char* p = ws + off;
    off = (off + bytes + 255) & ~(size_t)255;
    return p;
  };
  float*  npj      = (float*)alloc((size_t)V * KP * 4);
  __bf16* nf_bf    = (__bf16*)alloc((size_t)V * DD * 2);
  __bf16* WkB      = (__bf16*)alloc((size_t)(KKP / 32) * (OUTD / 16) * 512 * 2);
  __bf16* WnB      = (__bf16*)alloc((size_t)(DD / 32) * (DD / 16) * 512 * 2);
  __bf16* WihB     = (__bf16*)alloc((size_t)(DD / 32) * (768 / 16) * 512 * 2);
  __bf16* WhhB     = (__bf16*)alloc((size_t)(DD / 32) * (768 / 16) * 512 * 2);
  __bf16* WcB      = (__bf16*)alloc((size_t)(512 / 32) * (DD / 16) * 512 * 2);
  float*  kronF    = (float*)alloc((size_t)V * OUTD * 4);
  float*  s1       = (float*)alloc((size_t)V * 4);
  float*  s2       = (float*)alloc((size_t)V * 4);
  float*  mmax     = (float*)alloc((size_t)V * 4);
  float*  denom    = (float*)alloc((size_t)V * 4);
  float*  logit    = (float*)alloc((size_t)E * 4);
  float*  ex       = (float*)alloc((size_t)E * 4);
  float*  att      = (float*)alloc((size_t)E * 4);
  float*  hv       = (float*)alloc((size_t)V * DD * 4);
  float*  context  = (float*)alloc((size_t)V * DD * 4);
  __bf16* ctx_bf   = (__bf16*)alloc((size_t)V * DD * 2);
  __bf16* cat_bf   = (__bf16*)alloc((size_t)V * 512 * 2);
  float*  tmp      = (float*)alloc((size_t)V * DD * 4);
  const int mtilesV = V / 16;
  const int CT = 625;                      // GRU chunk (tiles of 16 nodes)
  float* gi_ch = (float*)alloc((size_t)CT * 16 * 768 * 4);
  float* gh_ch = (float*)alloc((size_t)CT * 16 * 768 * 4);
  (void)ws_size; (void)n_in; (void)out_size;

  auto gsz = [](long long n, int b) { return (unsigned)((n + b - 1) / b); };

  // ---- zero-init atomic targets ----
  k_zero_f32<<<gsz((long long)V * OUTD, 256), 256, 0, stream>>>(kronF, (long long)V * OUTD);
  k_zero_f32<<<gsz((long long)V * DD, 256), 256, 0, stream>>>(context, (long long)V * DD);
  k_zero_f32<<<gsz(V, 256), 256, 0, stream>>>(mmax, V);
  k_zero_f32<<<gsz(V, 256), 256, 0, stream>>>(denom, V);

  // ---- conversions + weight packing ----
  k_cvt_bf16<<<gsz((long long)V * DD, 256), 256, 0, stream>>>(nf, nf_bf, (long long)V * DD);
  k_pack_b<<<gsz((long long)(KKP / 32) * (OUTD / 16) * 512, 256), 256, 0, stream>>>(
      Wk, WkB, KK, KKP, OUTD, 0);
  k_pack_b<<<gsz((long long)(DD / 32) * (DD / 16) * 512, 256), 256, 0, stream>>>(
      Wn, WnB, DD, DD, DD, 0);
  k_pack_b<<<gsz((long long)(DD / 32) * (768 / 16) * 512, 256), 256, 0, stream>>>(
      W_ih, WihB, DD, DD, 768, 1);
  k_pack_b<<<gsz((long long)(DD / 32) * (768 / 16) * 512, 256), 256, 0, stream>>>(
      W_hh, WhhB, DD, DD, 768, 1);
  k_pack_b<<<gsz((long long)(512 / 32) * (DD / 16) * 512, 256), 256, 0, stream>>>(
      Wc, WcB, 512, 512, DD, 0);

  // ---- Kronecker branch ----
  k_node_proj<<<gsz(V, 16), 320, 0, stream>>>(nf, Wp, bp, gp, betap, npj, V);
  k_edge_kron<<<gsz(E, 16), 256, 0, stream>>>(npj, src, dst, WkB, bk, gk, betak,
                                              kronF, E);

  // ---- attention ----
  k_node_s12<<<gsz((long long)V * 32, 256), 256, 0, stream>>>(nf, We, s1, s2, V);
  k_edge_logit_max<<<gsz(E, 256), 256, 0, stream>>>(s1, s2, be, src, dst, logit,
                                                    mmax, E);
  k_edge_exp_sum<<<gsz(E, 256), 256, 0, stream>>>(logit, mmax, dst, ex, denom, E);
  k_edge_att<<<gsz(E, 256), 256, 0, stream>>>(ex, denom, dst, att, E);

  // ---- hv = nf @ Wn + bn  (WMMA) ----
  k_gemm_bf16<2><<<mtilesV, 256, 0, stream>>>(nf_bf, DD, WnB, DD, bn, hv);

  // ---- weighted aggregation -> context ----
  k_context_agg<<<gsz((long long)E * 64, 256), 256, 0, stream>>>(hv, att, src,
                                                                 dst, context, E);
  k_relu_cvt_bf16<<<gsz((long long)V * DD, 256), 256, 0, stream>>>(
      context, ctx_bf, (long long)V * DD);

  // ---- GRU: gi/gh GEMMs (WMMA, chunked) + gates + LN -> cat_bf ----
  for (int t0 = 0; t0 < mtilesV; t0 += CT) {
    int mt = (mtilesV - t0 < CT) ? (mtilesV - t0) : CT;
    int nodeBase = t0 * 16, nNodes = mt * 16;
    k_gemm_bf16<6><<<mt, 256, 0, stream>>>(ctx_bf + (long long)nodeBase * DD, DD,
                                           WihB, 768, b_ih, gi_ch);
    k_gemm_bf16<6><<<mt, 256, 0, stream>>>(nf_bf + (long long)nodeBase * DD, DD,
                                           WhhB, 768, b_hh, gh_ch);
    k_gru_gates<<<gsz((long long)nNodes * 32, 256), 256, 0, stream>>>(
        gi_ch, gh_ch, nf, kronF, g_ln, b_ln, cat_bf, nodeBase, nNodes);
  }

  // ---- concat projection + final LN + relu ----
  k_gemm_bf16<2><<<mtilesV, 256, 0, stream>>>(cat_bf, 512, WcB, DD, bc, tmp);
  k_ln_relu_out<<<gsz((long long)V * 32, 256), 256, 0, stream>>>(
      tmp, gc, betac, (float*)d_out, V);
}
